// ConstQAtt_2362232013127
// MI455X (gfx1250) — compile-verified, matrix operand order checked by reading, not run on hardware
//
#include <hip/hip_runtime.h>
#include <hip/hip_bf16.h>

// Problem constants (from reference)
constexpr int B = 4;
constexpr int S = 8192;
constexpr int D = 1024;   // DIM
constexpr int H = 1024;   // HDIM
constexpr int T = 64;     // NUM_TOKENS

typedef __bf16 bf16_t;
typedef __attribute__((ext_vector_type(8)))  __bf16 v8bf;
typedef __attribute__((ext_vector_type(16))) __bf16 v16bf;
typedef __attribute__((ext_vector_type(8)))  float  v8f;
typedef __attribute__((ext_vector_type(4)))  unsigned int v4u;
typedef __attribute__((ext_vector_type(8)))  int v8i;
typedef __attribute__((ext_vector_type(4)))  int v4i;

__device__ __forceinline__ v16bf cat16(v8bf lo, v8bf hi) {
    return __builtin_shufflevector(lo, hi, 0,1,2,3,4,5,6,7,8,9,10,11,12,13,14,15);
}

// ---------------------------------------------------------------------------
// CDNA5 async global->LDS DMA (ASYNCcnt-tracked).  INST_OFFSET is added to
// BOTH the global and LDS addresses, so one (lds, global) pair moves 64B/lane.
// ---------------------------------------------------------------------------
__device__ __forceinline__ void async_ld64B(unsigned int lds_off,
                                            unsigned long long gaddr) {
    asm volatile(
        "global_load_async_to_lds_b128 %0, %1, off\n"
        "global_load_async_to_lds_b128 %0, %1, off offset:16\n"
        "global_load_async_to_lds_b128 %0, %1, off offset:32\n"
        "global_load_async_to_lds_b128 %0, %1, off offset:48\n"
        :: "v"(lds_off), "v"(gaddr) : "memory");
}
__device__ __forceinline__ void wait_async0() {
    asm volatile("s_wait_asynccnt 0x0" ::: "memory");
}

// ---------------------------------------------------------------------------
// Kernel 1: qk[t][d] = (1/sqrt(H)) * sum_h q[t,h] * wk[h,d]   -> bf16
// ---------------------------------------------------------------------------
__global__ __launch_bounds__(256) void qk_kernel(const float* __restrict__ q,
                                                 const float* __restrict__ wk,
                                                 bf16_t* __restrict__ qk) {
    int id = blockIdx.x * 256 + threadIdx.x;     // 0 .. T*D-1
    int t = id >> 10;
    int d = id & (D - 1);
    float acc = 0.0f;
    for (int hh = 0; hh < H; ++hh)
        acc += q[t * H + hh] * wk[hh * D + d];
    qk[id] = (bf16_t)(acc * 0.03125f);           // 1/sqrt(1024)
}

// ---------------------------------------------------------------------------
// Kernel 2: scores[b,t,s] = sum_d qk[t,d] * x[b,s,d]
// TDM-staged: one wave issues a 2D tensor_load_to_lds descriptor per K-step
// (tile 32 x 128 f32, tensor stride D, HW padding 32->36 dwords/row),
// double-buffered and synced with s_wait_tensorcnt + workgroup barrier.
// ---------------------------------------------------------------------------
__global__ __launch_bounds__(256) void scores_kernel(const float* __restrict__ x,
                                                     const bf16_t* __restrict__ qk,
                                                     float* __restrict__ scores) {
    __shared__ float xsf[2][128][36];            // [buf][s_local][k_local], HW-padded

    const int b   = blockIdx.x >> 6;             // / (S/128)
    const int s0  = (blockIdx.x & 63) << 7;      // *128
    const int tid = threadIdx.x;
    const int wave = tid >> 5;
    const int lane = tid & 31;
    const int h = lane >> 4;                     // lane half
    const int n = lane & 15;                     // N index / A row-within-tile

    const float* xb = x + ((size_t)b * S) * D;
    const float* gtile = xb + (size_t)s0 * D;    // tile row origin

    // D# group1 (constant across steps): data_size=4B, pad 32dw interval (enc 4),
    // pad 4dw amount (enc 3), tensor_dim0=1024, tensor_dim1=8192,
    // tile_dim0=32, tile_dim1=128, tensor_dim0_stride=1024.
    v8i g1;
    g1[0] = (2 << 16) | (1 << 20) | (4 << 22) | (3 << 25);
    g1[1] = (int)(1024u << 16);                  // tensor_dim0[15:0]
    g1[2] = (int)(8192u << 16);                  // tensor_dim0[31:16]=0 | tensor_dim1[15:0]
    g1[3] = (int)(32u << 16);                    // tensor_dim1[31:16]=0 | tile_dim0=32
    g1[4] = 128;                                 // tile_dim1=128, tile_dim2=0
    g1[5] = 1024;                                // tensor_dim0_stride[31:0]
    g1[6] = 0;
    g1[7] = 0;
    const v4i zg  = {0, 0, 0, 0};
    const v8i zg8 = {0, 0, 0, 0, 0, 0, 0, 0};

    auto issue_tdm = [&](int bufsel, int kb) {
        unsigned long long ga = (unsigned long long)(uintptr_t)(gtile + kb);
        v4u g0;
        g0.x = 1u;                               // count=1 (valid descriptor)
        g0.y = (unsigned int)(uintptr_t)&xsf[bufsel][0][0];      // lds_addr
        g0.z = (unsigned int)ga;                                 // global_addr[31:0]
        g0.w = ((unsigned int)(ga >> 32) & 0x01FFFFFFu) | (2u << 30); // addr[56:32]|type=2
        __builtin_amdgcn_tensor_load_to_lds(g0, g1, zg, zg, zg8, 0);
    };

    v8f acc[4] = {};

    if (tid < 32) issue_tdm(0, 0);               // prologue: DMA K-step 0

    for (int i = 0; i < D / 32; ++i) {
        if (tid < 32) __builtin_amdgcn_s_wait_tensorcnt(0);
        __syncthreads();                         // slice i visible to all waves;
                                                 // slice i-1 reads complete
        if (i + 1 < D / 32 && tid < 32)          // overlap DMA of i+1 w/ compute
            issue_tdm((i + 1) & 1, (i + 1) * 32);

        const int kb = i * 32;
        const float (*buf)[36] = xsf[i & 1];
        const int sw = wave * 16 + n;

        // B fragment: B[k][n] = x[b, s0+sw, kb+k]; k = 16*(e/8)+8h+(e%8)
        const float4* rowp = (const float4*)&buf[sw][0];
        float4 f0 = rowp[2 * h];
        float4 f1 = rowp[2 * h + 1];
        float4 f2 = rowp[4 + 2 * h];
        float4 f3 = rowp[4 + 2 * h + 1];
        v16bf bfrag;
        bfrag[0]  = (bf16_t)f0.x; bfrag[1]  = (bf16_t)f0.y;
        bfrag[2]  = (bf16_t)f0.z; bfrag[3]  = (bf16_t)f0.w;
        bfrag[4]  = (bf16_t)f1.x; bfrag[5]  = (bf16_t)f1.y;
        bfrag[6]  = (bf16_t)f1.z; bfrag[7]  = (bf16_t)f1.w;
        bfrag[8]  = (bf16_t)f2.x; bfrag[9]  = (bf16_t)f2.y;
        bfrag[10] = (bf16_t)f2.z; bfrag[11] = (bf16_t)f2.w;
        bfrag[12] = (bf16_t)f3.x; bfrag[13] = (bf16_t)f3.y;
        bfrag[14] = (bf16_t)f3.z; bfrag[15] = (bf16_t)f3.w;

#pragma unroll
        for (int mt = 0; mt < 4; ++mt) {
            const bf16_t* arow = qk + (size_t)(mt * 16 + n) * D + kb;
            v8bf alo = *(const v8bf*)(arow + 8 * h);
            v8bf ahi = *(const v8bf*)(arow + 16 + 8 * h);
            v16bf afrag = cat16(alo, ahi);
            acc[mt] = __builtin_amdgcn_wmma_f32_16x16x32_bf16(
                false, afrag, false, bfrag, (short)0, acc[mt], false, false);
        }
    }

    // D layout: VGPR r <-> M = r + 8*h, N = n
    const int scol = s0 + wave * 16 + n;
#pragma unroll
    for (int mt = 0; mt < 4; ++mt)
#pragma unroll
        for (int rr = 0; rr < 8; ++rr) {
            int t = mt * 16 + 8 * h + rr;
            scores[((size_t)(b * T + t)) * S + scol] = acc[mt][rr];
        }
}

// ---------------------------------------------------------------------------
// Kernel 3: softmax over s (adds broadcast mask), writes bf16 probs.
// ---------------------------------------------------------------------------
__global__ __launch_bounds__(256) void softmax_kernel(const float* __restrict__ scores,
                                                      const float* __restrict__ mask,
                                                      bf16_t* __restrict__ probs) {
    const int bt = blockIdx.x;                   // 0..B*T-1
    const int b  = bt >> 6;
    const float* row  = scores + (size_t)bt * S;
    const float* mrow = mask + (size_t)b * S;

    float v[32];
    float mx = -INFINITY;
#pragma unroll
    for (int k = 0; k < 32; ++k) {
        int i = threadIdx.x + k * 256;
        v[k] = row[i] + mrow[i];
        mx = fmaxf(mx, v[k]);
    }

    __shared__ float red[256];
    red[threadIdx.x] = mx;
    __syncthreads();
    for (int off = 128; off > 0; off >>= 1) {
        if (threadIdx.x < off)
            red[threadIdx.x] = fmaxf(red[threadIdx.x], red[threadIdx.x + off]);
        __syncthreads();
    }
    mx = red[0];
    __syncthreads();

    float sum = 0.0f;
#pragma unroll
    for (int k = 0; k < 32; ++k) {
        v[k] = __expf(v[k] - mx);
        sum += v[k];
    }
    red[threadIdx.x] = sum;
    __syncthreads();
    for (int off = 128; off > 0; off >>= 1) {
        if (threadIdx.x < off)
            red[threadIdx.x] += red[threadIdx.x + off];
        __syncthreads();
    }
    const float inv = 1.0f / red[0];

#pragma unroll
    for (int k = 0; k < 32; ++k) {
        int i = threadIdx.x + k * 256;
        probs[(size_t)bt * S + i] = (bf16_t)(v[k] * inv);
    }
}

// ---------------------------------------------------------------------------
// Kernel 4: u[b,t,d] = sum_s probs[b,t,s] * x[b,s,d]
// Async double-buffered f32 staging (untransposed [32 s][132 d]); B-fragment
// columns read per-element (lanes n=0..15 hit consecutive floats of a row:
// conflict-free), converted to bf16 at build.
// ---------------------------------------------------------------------------
__global__ __launch_bounds__(256) void pv_kernel(const float* __restrict__ x,
                                                 const bf16_t* __restrict__ probs,
                                                 float* __restrict__ u) {
    __shared__ float xsf[2][32][132];            // [buf][s_local][d_local], padded

    const int b   = blockIdx.x >> 3;             // / (D/128)
    const int d0  = (blockIdx.x & 7) << 7;       // *128
    const int tid = threadIdx.x;
    const int wave = tid >> 5;
    const int lane = tid & 31;
    const int h = lane >> 4;
    const int n = lane & 15;

    const float* xb = x + ((size_t)b * S) * D;
    const bf16_t* pb = probs + (size_t)(b * T) * S;

    // staging: thread moves 16 contiguous floats of row sl, col chunk c
    const int sl = tid >> 3;                     // 0..31
    const int c  = (tid & 7) * 16;               // 0..112
    const unsigned int lds0 = (unsigned int)(uintptr_t)&xsf[0][sl][c];
    const unsigned int lds1 = (unsigned int)(uintptr_t)&xsf[1][sl][c];
    const float* grow = xb + (size_t)sl * D + d0 + c;

    v8f acc[4] = {};

    // prologue: DMA K-step 0 into buffer 0
    async_ld64B(lds0, (unsigned long long)(uintptr_t)grow);

    for (int i = 0; i < S / 32; ++i) {
        wait_async0();
        __syncthreads();
        if (i + 1 < S / 32) {
            unsigned int dst = ((i + 1) & 1) ? lds1 : lds0;
            async_ld64B(dst, (unsigned long long)(uintptr_t)
                                 (grow + (size_t)(i + 1) * 32 * D));
        }

        const int kb = i * 32;
        const float (*buf)[132] = xsf[i & 1];
        const int dl = wave * 16 + n;

        // B fragment: B[k][n] = x[b, kb+k, d0+dl]; k = 16*(e/8)+8h+(e%8)
        v16bf bfrag;
#pragma unroll
        for (int e = 0; e < 8; ++e) {
            bfrag[e]     = (bf16_t)buf[8 * h + e][dl];
            bfrag[8 + e] = (bf16_t)buf[16 + 8 * h + e][dl];
        }

#pragma unroll
        for (int mt = 0; mt < 4; ++mt) {
            const bf16_t* arow = pb + (size_t)(mt * 16 + n) * S + kb;
            v8bf alo = *(const v8bf*)(arow + 8 * h);
            v8bf ahi = *(const v8bf*)(arow + 16 + 8 * h);
            v16bf afrag = cat16(alo, ahi);
            acc[mt] = __builtin_amdgcn_wmma_f32_16x16x32_bf16(
                false, afrag, false, bfrag, (short)0, acc[mt], false, false);
        }
    }

    const int dcol = d0 + wave * 16 + n;
#pragma unroll
    for (int mt = 0; mt < 4; ++mt)
#pragma unroll
        for (int r = 0; r < 8; ++r) {
            int t = mt * 16 + 8 * h + r;
            u[((size_t)(b * T + t)) * D + dcol] = acc[mt][r];
        }
}

// ---------------------------------------------------------------------------
// Kernel 5: out[b,t,h] = sum_d u[b,t,d] * wv[h,d]
// ---------------------------------------------------------------------------
__global__ __launch_bounds__(256) void out_kernel(const float* __restrict__ u,
                                                  const float* __restrict__ wv,
                                                  float* __restrict__ out) {
    int id = blockIdx.x * 256 + threadIdx.x;     // 0 .. B*T*H-1
    int bt = id >> 10;
    int hc = id & (H - 1);
    const float* ur = u + (size_t)bt * D;
    const float* wr = wv + (size_t)hc * D;
    float acc = 0.0f;
    for (int d = 0; d < D; d += 4) {
        acc += ur[d + 0] * wr[d + 0];
        acc += ur[d + 1] * wr[d + 1];
        acc += ur[d + 2] * wr[d + 2];
        acc += ur[d + 3] * wr[d + 3];
    }
    out[id] = acc;
}

// ---------------------------------------------------------------------------
extern "C" void kernel_launch(void* const* d_in, const int* in_sizes, int n_in,
                              void* d_out, int out_size, void* d_ws, size_t ws_size,
                              hipStream_t stream) {
    const float* x    = (const float*)d_in[0];   // (B,S,D)
    const float* mask = (const float*)d_in[1];   // (B,1,S)
    const float* wk   = (const float*)d_in[2];   // (H,D)
    const float* wv   = (const float*)d_in[3];   // (H,D)
    const float* q    = (const float*)d_in[4];   // (T,H)
    float* out = (float*)d_out;                  // (B,T,H)

    char* ws = (char*)d_ws;
    bf16_t* qk_ws    = (bf16_t*)(ws);                                  // T*D bf16   = 128 KB
    float*  score_ws = (float*)(ws + 131072);                          // B*T*S f32  = 8 MB
    bf16_t* probs_ws = (bf16_t*)(ws + 131072 + 8388608);               // B*T*S bf16 = 4 MB
    float*  u_ws     = (float*)(ws + 131072 + 8388608 + 4194304);      // B*T*D f32  = 1 MB

    qk_kernel<<<(T * D) / 256, 256, 0, stream>>>(q, wk, qk_ws);
    scores_kernel<<<B * (S / 128), 256, 0, stream>>>(x, qk_ws, score_ws);
    softmax_kernel<<<B * T, 256, 0, stream>>>(score_ws, mask, probs_ws);
    pv_kernel<<<B * (D / 128), 256, 0, stream>>>(x, probs_ws, u_ws);
    out_kernel<<<(B * T * H) / 256, 256, 0, stream>>>(u_ws, wv, out);
}
